// SubLSTM_64072322122357
// MI455X (gfx1250) — matrix operand order repaired and checked by reading, not compiled, not run on hardware
//
#include <hip/hip_runtime.h>
#include <hip/hip_bf16.h>

// subLSTM for MI455X (gfx1250, wave32, WMMA).
// Fused-K recurrent GEMM: A = [x_t | h_{t-1}] (bf16, K = 2048),
// B = [W_in ; W_rec] rows (bf16), f32 accumulate via v_wmma_f32_16x16x32_bf16.
// One wave owns a full 64-row x 16-col output slice across ALL FOUR gates:
// 16 accumulators, 16 WMMA per K-step fed by 16 b128 loads (1.0 load/WMMA,
// zero intra-wave duplication). K-loop is software-pipelined (cur/next frag
// sets) so loads of step k+1 overlap the 16-WMMA burst of step k.
// Weights (16.8 MB bf16) + h + x_t stay L2-resident across all 512 steps.

#define T_STEPS 512
#define BATCH   64
#define IN_DIM  1024
#define HID     1024
#define GDIM    (4 * HID)
#define KTOT    (IN_DIM + HID)   // 2048

typedef __attribute__((ext_vector_type(16))) __bf16 v16bf;
typedef __attribute__((ext_vector_type(8)))  __bf16 v8bf;
typedef __attribute__((ext_vector_type(8)))  float  v8f;

__device__ __forceinline__ float sigmoidf_(float x) {
    return 1.0f / (1.0f + __expf(-x));
}

__device__ __forceinline__ v16bf cat16(v8bf lo, v8bf hi) {
    return __builtin_shufflevector(lo, hi,
        0, 1, 2, 3, 4, 5, 6, 7, 8, 9, 10, 11, 12, 13, 14, 15);
}

// ---------- pre-pass: fp32 -> bf16 conversions ----------

__global__ void k_convert_x(const float* __restrict__ x, __bf16* __restrict__ xbf, int n) {
    int i = 4 * (blockIdx.x * blockDim.x + threadIdx.x);
    if (i + 3 < n) {
        float4 v = *(const float4*)(x + i);
        xbf[i + 0] = (__bf16)v.x;
        xbf[i + 1] = (__bf16)v.y;
        xbf[i + 2] = (__bf16)v.z;
        xbf[i + 3] = (__bf16)v.w;
    }
}

// Wcat[g][k] bf16: k<I -> W_in[g][k], else W_rec[g][k-I]. Row-major, stride KTOT.
__global__ void k_build_wcat(const float* __restrict__ W_in,
                             const float* __restrict__ W_rec,
                             __bf16* __restrict__ wcat) {
    int i = 4 * (blockIdx.x * blockDim.x + threadIdx.x);
    if (i + 3 < GDIM * KTOT) {
        int g = i / KTOT;
        int k = i - g * KTOT;
        const float* src = (k < IN_DIM) ? (W_in + (size_t)g * IN_DIM + k)
                                        : (W_rec + (size_t)g * HID + (k - IN_DIM));
        float4 v = *(const float4*)src;
        wcat[i + 0] = (__bf16)v.x;
        wcat[i + 1] = (__bf16)v.y;
        wcat[i + 2] = (__bf16)v.z;
        wcat[i + 3] = (__bf16)v.w;
    }
}

__global__ void k_init_state(const float* __restrict__ h0, const float* __restrict__ c0,
                             __bf16* __restrict__ hbf0, float* __restrict__ cbuf, int n) {
    int i = blockIdx.x * blockDim.x + threadIdx.x;
    if (i < n) {
        cbuf[i] = c0[i];
        hbf0[i] = (__bf16)h0[i];
    }
}

// ---------- one recurrent step ----------
// Grid: HID/16 = 64 blocks of one wave (32 threads).
// Wave owns output cols [blockIdx*16, +16) for all 4 gates and all 64 rows.
//
// WMMA bf16 fragment layouts (CDNA5 ISA 7.12.2):
//   A 16x32: lane L(0..15): M=L, K = {0..7, 16..23}; lane L+16: K = {8..15, 24..31}
//            -> two contiguous 8-elem (16B) runs at K = 8*laneHi and 16 + 8*laneHi.
//   B 32x16: lane L: N=L%16, K same as A -> two 8-elem (16B) runs at K = 8*laneHi and
//            16 + 8*laneHi of row-major Wcat (row = output column).
//   C/D 16x16 f32: VGPR r: M = r + 8*laneHi, N = lane%16.

struct Frag {
    v8bf a[4][2];   // A fragments for m2 = 0..3
    v8bf b[4][2];   // B fragments for gate = 0..3
};

__device__ __forceinline__ Frag load_set(const __bf16* __restrict__ abase, int ka,
                                         const __bf16* __restrict__ w0,
                                         const __bf16* __restrict__ w1,
                                         const __bf16* __restrict__ w2,
                                         const __bf16* __restrict__ w3, int kb) {
    Frag f;
#pragma unroll
    for (int m2 = 0; m2 < 4; ++m2) {
        const __bf16* p = abase + (size_t)m2 * (16 * 1024) + ka;
        f.a[m2][0] = *(const v8bf*)(p);
        f.a[m2][1] = *(const v8bf*)(p + 16);
    }
    f.b[0][0] = *(const v8bf*)(w0 + kb); f.b[0][1] = *(const v8bf*)(w0 + kb + 16);
    f.b[1][0] = *(const v8bf*)(w1 + kb); f.b[1][1] = *(const v8bf*)(w1 + kb + 16);
    f.b[2][0] = *(const v8bf*)(w2 + kb); f.b[2][1] = *(const v8bf*)(w2 + kb + 16);
    f.b[3][0] = *(const v8bf*)(w3 + kb); f.b[3][1] = *(const v8bf*)(w3 + kb + 16);
    return f;
}

__device__ __forceinline__ void mma_set(const Frag& f, v8f (&acc)[4][4]) {
#pragma unroll
    for (int g = 0; g < 4; ++g) {
        const v16bf bg = cat16(f.b[g][0], f.b[g][1]);
#pragma unroll
        for (int m2 = 0; m2 < 4; ++m2) {
            const v16bf am = cat16(f.a[m2][0], f.a[m2][1]);
            acc[g][m2] = __builtin_amdgcn_wmma_f32_16x16x32_bf16(
                false, am, false, bg, (short)0, acc[g][m2], false, false);
        }
    }
}

__global__ __launch_bounds__(32) void k_sublstm_step(
    const __bf16* __restrict__ xbf,    // [T][B][I]
    const __bf16* __restrict__ wcat,   // [G][KTOT]
    const float*  __restrict__ bias,   // [G]
    const __bf16* __restrict__ hbf_in, // [B][H]
    __bf16*       __restrict__ hbf_out,
    float*        __restrict__ cbuf,   // [B][H]
    float*        __restrict__ outs,   // [T][B][H] (head of d_out)
    int t)
{
    const int lane   = threadIdx.x & 31;
    const int laneHi = lane >> 4;
    const int l16    = lane & 15;

    const int hcol = blockIdx.x * 16 + l16;   // this lane's output / B-row column

    // A bases: batch row = l16 (+ m2*16 inside load_set); x and h share stride 1024.
    const __bf16* xa = xbf + ((size_t)t * BATCH + l16) * IN_DIM;
    const __bf16* ha = hbf_in + (size_t)l16 * HID;

    const __bf16* w0 = wcat + (size_t)(0 * HID + hcol) * KTOT;
    const __bf16* w1 = wcat + (size_t)(1 * HID + hcol) * KTOT;
    const __bf16* w2 = wcat + (size_t)(2 * HID + hcol) * KTOT;
    const __bf16* w3 = wcat + (size_t)(3 * HID + hcol) * KTOT;

    const int ka = 8 * laneHi;    // A run offset within 32-wide K tile
    const int kb = 8 * laneHi;    // B run offset within 32-wide K tile

    v8f acc[4][4] = {};           // [gate][m2]

    // ---- software-pipelined K loop: x segment then h segment ----
    Frag cur = load_set(xa, ka, w0, w1, w2, w3, kb);

#pragma unroll 2
    for (int k0 = 0; k0 < IN_DIM - 32; k0 += 32) {
        Frag nxt = load_set(xa, k0 + 32 + ka, w0, w1, w2, w3, k0 + 32 + kb);
        mma_set(cur, acc);
        cur = nxt;
    }
    {   // segment boundary: prefetch first h fragment, finish last x fragment
        Frag nxt = load_set(ha, ka, w0, w1, w2, w3, IN_DIM + kb);
        mma_set(cur, acc);
        cur = nxt;
    }
#pragma unroll 2
    for (int k0 = 0; k0 < HID - 32; k0 += 32) {
        Frag nxt = load_set(ha, k0 + 32 + ka, w0, w1, w2, w3, IN_DIM + k0 + 32 + kb);
        mma_set(cur, acc);
        cur = nxt;
    }
    mma_set(cur, acc);

    // ---- fused subLSTM epilogue ----
    const float bi  = bias[0 * HID + hcol];
    const float bo  = bias[1 * HID + hcol];
    const float bz  = bias[2 * HID + hcol];
    const float bfv = bias[3 * HID + hcol];

    float* outs_t = outs + (size_t)t * BATCH * HID;

#pragma unroll
    for (int m2 = 0; m2 < 4; ++m2) {
#pragma unroll
        for (int r = 0; r < 8; ++r) {
            const int m = m2 * 16 + 8 * laneHi + r;
            const size_t idx = (size_t)m * HID + hcol;

            const float ig = sigmoidf_(acc[0][m2][r] + bi);
            const float og = sigmoidf_(acc[1][m2][r] + bo);
            const float zg = sigmoidf_(acc[2][m2][r] + bz);
            const float fg = sigmoidf_(acc[3][m2][r] + bfv);

            const float cold = cbuf[idx];
            const float cnew = cold * fg + zg - ig;   // subtractive cell update
            const float hnew = sigmoidf_(cnew) - og;  // subtractive output

            cbuf[idx]    = cnew;
            outs_t[idx]  = hnew;
            hbf_out[idx] = (__bf16)hnew;
        }
    }
}

// ---------- finalize: hT = outs[T-1], cT = cbuf ----------
__global__ void k_finalize(const float* __restrict__ cbuf, float* __restrict__ out) {
    int i = blockIdx.x * blockDim.x + threadIdx.x;
    const size_t outsN = (size_t)T_STEPS * BATCH * HID;
    if (i < BATCH * HID) {
        out[outsN + i]               = out[(size_t)(T_STEPS - 1) * BATCH * HID + i];
        out[outsN + BATCH * HID + i] = cbuf[i];
    }
}

extern "C" void kernel_launch(void* const* d_in, const int* in_sizes, int n_in,
                              void* d_out, int out_size, void* d_ws, size_t ws_size,
                              hipStream_t stream) {
    const float* x     = (const float*)d_in[0];   // [T,B,I]
    const float* h0    = (const float*)d_in[1];   // [B,H]
    const float* c0    = (const float*)d_in[2];   // [B,H]
    const float* W_in  = (const float*)d_in[3];   // [G,I]
    const float* bias  = (const float*)d_in[4];   // [G]
    const float* W_rec = (const float*)d_in[5];   // [G,H]
    float* out = (float*)d_out;

    // workspace layout (~84.5 MB total)
    char* ws = (char*)d_ws;
    size_t off = 0;
    __bf16* wcat = (__bf16*)(ws + off); off += (size_t)GDIM * KTOT * sizeof(__bf16);
    __bf16* xbf  = (__bf16*)(ws + off); off += (size_t)T_STEPS * BATCH * IN_DIM * sizeof(__bf16);
    float*  cbuf = (float*)(ws + off);  off += (size_t)BATCH * HID * sizeof(float);
    __bf16* hbf0 = (__bf16*)(ws + off); off += (size_t)BATCH * HID * sizeof(__bf16);
    __bf16* hbf1 = (__bf16*)(ws + off); off += (size_t)BATCH * HID * sizeof(__bf16);
    (void)ws_size;

    {
        const int n = T_STEPS * BATCH * IN_DIM;
        k_convert_x<<<(n / 4 + 255) / 256, 256, 0, stream>>>(x, xbf, n);
    }
    {
        const int n = GDIM * KTOT;
        k_build_wcat<<<(n / 4 + 255) / 256, 256, 0, stream>>>(W_in, W_rec, wcat);
    }
    k_init_state<<<(BATCH * HID + 255) / 256, 256, 0, stream>>>(h0, c0, hbf0, cbuf, BATCH * HID);

    __bf16* hin = hbf0;
    __bf16* hout = hbf1;
    for (int t = 0; t < T_STEPS; ++t) {
        k_sublstm_step<<<HID / 16, 32, 0, stream>>>(xbf, wcat, bias, hin, hout, cbuf, out, t);
        __bf16* tmp = hin; hin = hout; hout = tmp;
    }

    k_finalize<<<(BATCH * HID + 255) / 256, 256, 0, stream>>>(cbuf, out);
}
